// TorchSparseMultiTokenAttention_4784593568451
// MI455X (gfx1250) — matrix operand order repaired and checked by reading, not compiled, not run on hardware
//
#include <hip/hip_runtime.h>
#include <hip/hip_bf16.h>

// MI455X / gfx1250, wave32. Two-stage fused pipeline:
//   1) sparsemax_rows: register-resident row, bisection for tau (no sort),
//      single HBM read of scores, bf16 probs -> workspace (134 MB, fits the
//      192 MB global L2, so the conv's halo re-reads never touch HBM).
//   2) conv3x3_wmma: grouped 3x3 conv as implicit GEMM with a channel-fastest
//      im2col K ordering (K = (ky*3+kx)*16 + ci) so every WMMA B fragment is
//      one contiguous 32-byte LDS slice: 2x ds_load_b128 + 1 wmma per K-chunk.

typedef __attribute__((ext_vector_type(16))) __bf16          v16bf;
typedef __attribute__((ext_vector_type(8)))  float           v8f;
typedef __attribute__((ext_vector_type(8)))  unsigned short  u16x8;

#define LDIM 2048
#define NEG_INF_F (-1000000000.0f)

__device__ __forceinline__ unsigned short f2bf_bits(float f) {
  unsigned u = __float_as_uint(f);
  u = u + 0x7FFFu + ((u >> 16) & 1u);            // round-to-nearest-even
  return (unsigned short)(u >> 16);
}

struct bfpair { u16x8 lo, hi; };
__device__ __forceinline__ v16bf make_frag(u16x8 lo, u16x8 hi) {
  bfpair p{lo, hi};
  return __builtin_bit_cast(v16bf, p);           // 32B -> 16 bf16 fragment
}

__device__ __forceinline__ float wave_sum(float v) {
#pragma unroll
  for (int o = 16; o > 0; o >>= 1) v += __shfl_down(v, o, 32);
  return v;
}
__device__ __forceinline__ float wave_max(float v) {
#pragma unroll
  for (int o = 16; o > 0; o >>= 1) v = fmaxf(v, __shfl_down(v, o, 32));
  return v;
}

// ---------------------------------------------------------------------------
// Stage 1: causal-masked sparsemax per row. One 256-thread block (8 waves)
// per (channel,row); 8 f32 per thread keeps the whole row in VGPRs.
// ---------------------------------------------------------------------------
__global__ __launch_bounds__(256) void sparsemax_rows(
    const float* __restrict__ scores, unsigned short* __restrict__ probs) {
  __shared__ float red[9];
  const int row  = blockIdx.x;                 // c*L + y
  const int y    = row & (LDIM - 1);
  const int tid  = threadIdx.x;
  const int wave = tid >> 5;
  const int lane = tid & 31;
  const int base = tid * 8;

  const float* z = scores + (size_t)row * LDIM;
  float zv[8];
  const float4* zp = (const float4*)(z + base); // 32B-aligned b128 loads
  float4 a = zp[0], b = zp[1];
  zv[0]=a.x; zv[1]=a.y; zv[2]=a.z; zv[3]=a.w;
  zv[4]=b.x; zv[5]=b.y; zv[6]=b.z; zv[7]=b.w;
#pragma unroll
  for (int j = 0; j < 8; ++j) if (base + j > y) zv[j] = NEG_INF_F;

  // row max (support never empty: diagonal is always valid, > POS_THRESH)
  float m = zv[0];
#pragma unroll
  for (int j = 1; j < 8; ++j) m = fmaxf(m, zv[j]);
  m = wave_max(m);
  if (lane == 0) red[wave] = m;
  __syncthreads();
  if (tid == 0) {
    float r = red[0];
#pragma unroll
    for (int i = 1; i < 8; ++i) r = fmaxf(r, red[i]);
    red[8] = r;
  }
  __syncthreads();
  m = red[8];

  // bisection: f(tau)=sum relu(z-tau) is monotone; tau* in [m-1, m]
  float lo = m - 1.0f, hi = m;
  for (int it = 0; it < 26; ++it) {
    __syncthreads();                           // protect red[] reuse
    const float tau = 0.5f * (lo + hi);
    float s = 0.0f;
#pragma unroll
    for (int j = 0; j < 8; ++j) s += fmaxf(zv[j] - tau, 0.0f);
    s = wave_sum(s);
    if (lane == 0) red[wave] = s;
    __syncthreads();
    if (tid == 0) {
      float r = 0.0f;
#pragma unroll
      for (int i = 0; i < 8; ++i) r += red[i];
      red[8] = r;
    }
    __syncthreads();
    s = red[8];
    if (s > 1.0f) lo = tau; else hi = tau;
  }
  const float tau = 0.5f * (lo + hi);

  // pack probs to bf16 (masked region becomes exact 0, matching reference)
  unsigned short o16[8];
#pragma unroll
  for (int j = 0; j < 8; ++j)
    o16[j] = f2bf_bits(fmaxf(zv[j] - tau, 0.0f));
  uint4 pk;
  pk.x = (unsigned)o16[0] | ((unsigned)o16[1] << 16);
  pk.y = (unsigned)o16[2] | ((unsigned)o16[3] << 16);
  pk.z = (unsigned)o16[4] | ((unsigned)o16[5] << 16);
  pk.w = (unsigned)o16[6] | ((unsigned)o16[7] << 16);
  *(uint4*)(probs + (size_t)row * LDIM + base) = pk;
}

// ---------------------------------------------------------------------------
// Stage 2: grouped 3x3 conv as implicit GEMM, K = (ky*3+kx)*16 + ci, padded
// 144 -> 160 (5 chunks of 32). Block = 256 thr (8 waves) covers one y and
// 128 x; each wave does a 16(c_out) x 16(x) tile.
//   sP2[ry][xx][ci] : channel-interleaved halo tile -> B fragments are
//                     contiguous 32B LDS slices (2x ds_load_b128 each).
//   sA[co][K]       : dense block-diagonal weights -> A fragments are
//                     loop-invariant contiguous loads (10x ds_load_b128).
// ---------------------------------------------------------------------------
#define CI_PAD 24   // 16 ci + pad; 48B stride keeps 16B alignment for b128

__global__ __launch_bounds__(256) void conv3x3_wmma(
    const unsigned short* __restrict__ probs, const float* __restrict__ weight,
    const float* __restrict__ bias, float* __restrict__ out) {
  __shared__ __align__(16) unsigned short sP2[3][132][CI_PAD];
  __shared__ __align__(16) unsigned short sA[16][160];
  __shared__ __align__(16) unsigned short sZero[16];
  __shared__ float sB[16];

  const int y   = blockIdx.y;
  const int x0  = blockIdx.x * 128;
  const int tid = threadIdx.x;

  // dense block-diagonal 16x160 bf16 weight matrix (zeros across groups/pad)
  for (int i = tid; i < 16 * 160; i += 256) {
    const int co = i / 160;
    const int K  = i - co * 160;
    const int k9 = K >> 4;                     // ky*3+kx
    const int ci = K & 15;
    float wv = 0.0f;
    if (k9 < 9 && (ci >> 2) == (co >> 2))
      wv = weight[(co * 4 + (ci & 3)) * 9 + k9];
    sA[co][K] = f2bf_bits(wv);
  }
  if (tid < 16) { sZero[tid] = 0; sB[tid] = bias[tid]; }

  // halo tile: rows y-1..y+1, x0-1..x0+128, all 16 channels (zero padded)
  for (int i = tid; i < 16 * 3 * 130; i += 256) {
    const int ci  = i / 390;
    const int rem = i - ci * 390;
    const int ry  = rem / 130;
    const int xx  = rem - ry * 130;
    const int gx  = x0 - 1 + xx;
    const int gy  = y - 1 + ry;
    unsigned short v = 0;
    if (gx >= 0 && gx < LDIM && gy >= 0 && gy < LDIM)
      v = probs[((size_t)ci * LDIM + gy) * LDIM + gx];
    sP2[ry][xx][ci] = v;
  }
  __syncthreads();

  const int wave = tid >> 5;
  const int lane = tid & 31;
  const int N    = lane & 15;                  // B/D column (pixel in tile)
  const int hi16 = lane >> 4;                  // half-wave select
  const int M    = N;                          // A row (c_out) for this lane

  // A fragments: lane<16 holds K kc*32+0..15; lane>=16 holds kc*32+16..31.
  const unsigned short* aRow = &sA[M][0];
  v16bf A[5];
#pragma unroll
  for (int kc = 0; kc < 5; ++kc) {
    const u16x8 alo = *(const u16x8*)(aRow + kc * 32 + hi16 * 16);
    const u16x8 ahi = *(const u16x8*)(aRow + kc * 32 + hi16 * 16 + 8);
    A[kc] = make_frag(alo, ahi);
  }

  // B fragments: half-wave (hi16) covers k9 = 2*kc + hi16; its 16 elements
  // are ci=0..15 at pixel xx = wave*16 + N + kx -> one contiguous 32B slice.
  const unsigned short* sPbase = &sP2[0][0][0];
  const int pixOff = (wave * 16 + N) * CI_PAD; // loop-invariant lane offset
  v8f c = {};
#pragma unroll
  for (int kc = 0; kc < 5; ++kc) {
    const int k9a  = 2 * kc, k9b = 2 * kc + 1;
    const int offa = ((k9a / 3) * 132 + (k9a % 3)) * CI_PAD;
    const int offb = ((k9b / 3) * 132 + (k9b % 3)) * CI_PAD;
    const unsigned short* bp;
    if (kc == 4) {                             // k9=9 pad -> zeros
      bp = hi16 ? &sZero[0] : (sPbase + offa + pixOff);
    } else {
      bp = sPbase + (hi16 ? offb : offa) + pixOff;
    }
    const v16bf b = make_frag(*(const u16x8*)bp, *(const u16x8*)(bp + 8));
    c = __builtin_amdgcn_wmma_f32_16x16x32_bf16(
        false, A[kc], false, b, (short)0, c, false, false);
  }

  // D layout: lane<16 -> M=r, lane>=16 -> M=8+r; col = lane&15
  const int x = x0 + wave * 16 + N;
#pragma unroll
  for (int r = 0; r < 8; ++r) {
    const int co = r + hi16 * 8;
    float o = c[r] + sB[co];
    if (x > y) o = 0.0f;                       // causal re-mask
    out[((size_t)co * LDIM + y) * LDIM + x] = o;
  }
}

extern "C" void kernel_launch(void* const* d_in, const int* in_sizes, int n_in,
                              void* d_out, int out_size, void* d_ws, size_t ws_size,
                              hipStream_t stream) {
  (void)in_sizes; (void)n_in; (void)out_size; (void)ws_size;
  const float* scores = (const float*)d_in[0];   // (1,16,2048,2048) f32
  const float* weight = (const float*)d_in[1];   // (16,4,3,3) f32
  const float* bias   = (const float*)d_in[2];   // (16,) f32
  float* out = (float*)d_out;                    // (1,16,2048,2048) f32
  unsigned short* probs = (unsigned short*)d_ws; // 16*2048*2048 bf16 = 128 MiB

  sparsemax_rows<<<16 * LDIM, 256, 0, stream>>>(scores, probs);
  dim3 grid(LDIM / 128, LDIM);
  conv3x3_wmma<<<grid, 256, 0, stream>>>(probs, weight, bias, out);
}
